// GreedyDecoder_13795434954802
// MI455X (gfx1250) — compile-verified, matrix operand order checked by reading, not run on hardware
//
#include <hip/hip_runtime.h>
#include <hip/hip_bf16.h>
#include <math.h>

// ---------------------------------------------------------------------------
// RNN-T greedy decoder for MI455X (gfx1250).
//  * persistent single-workgroup kernel (1024 thr = 32 wave32, 1 WGP)
//  * bf16 V_WMMA_F32_16X16X32_BF16 for all GEMMs, fp32 accumulate
//  * weights pre-swizzled into ISA 16x16x32 tile layout (prep kernel)
//  * LDS (160KB of the 320KB WGP pool) holds packed activation A-tiles
//  * operand streams hoisted to per-stream pointers + constant offsets so
//    global/ds loads use immediate offsets (no per-iteration address VALU)
// ---------------------------------------------------------------------------

typedef __bf16 bf16;
typedef __attribute__((ext_vector_type(16))) __bf16 v16bf;
typedef __attribute__((ext_vector_type(8)))  float  v8f;

#define T_        300
#define N_        64
#define C_        1024
#define H_        320
#define V_        29
#define JH_       512
#define BLANK_    28
#define SOS_      28
#define MAX_SYM_  2
#define MAX_STEPS_ 900
#define MAX_RES_  601

#define KT_L 20                 // K-tiles LSTM GEMM (K = 640)
#define KT_J 42                 // K-tiles joint fc1 (K = 1344)
#define KT_2 16                 // K-tiles fc2      (K = 512)
#define WL_ELEMS (80 * KT_L * 512)   // 1600 tiles * 512 = 819200 bf16 / layer

// Swizzle (row, k) -> flat index in 16(N/M) x 32(K) bf16 WMMA tile stream.
// Matches CDNA5 ISA "16-bit A-Matrix 16x32" layout:
//   lane = (row%16) + 16*((k%32)>>3 & 1),  elem = (k&7) + 8*(k%32 >= 16)
__device__ __host__ __forceinline__ int swz(int row, int k, int ktiles) {
    int nt = row >> 4;
    int kt = k >> 5;
    int v  = k & 31;
    int hi = (v >> 3) & 1;
    int e  = (v & 7) + ((v >> 4) << 3);
    int ln = (row & 15) + (hi << 4);
    return ((nt * ktiles + kt) << 9) + (ln << 4) + e;
}

// ----------------------------- weight prep --------------------------------
__global__ void prep_kernel(const float* __restrict__ W_ih,
                            const float* __restrict__ W_hh,
                            const float* __restrict__ W_fg,
                            const float* __restrict__ W_out,
                            const float* __restrict__ emb,
                            bf16* __restrict__ wL,  bf16* __restrict__ wFG,
                            bf16* __restrict__ wOUT, bf16* __restrict__ embb)
{
    const int stride = gridDim.x * blockDim.x;
    const int g = blockIdx.x * blockDim.x + threadIdx.x;

    // fused LSTM weights: per layer, 1280 x 640  ([W_ih | W_hh])
    for (int i = g; i < 2 * 1280 * 640; i += stride) {
        int l = i / (1280 * 640);
        int r = i - l * (1280 * 640);
        int n = r / 640;
        int k = r - n * 640;
        float v = (k < H_) ? W_ih[(l * 1280 + n) * H_ + k]
                           : W_hh[(l * 1280 + n) * H_ + (k - H_)];
        wL[l * WL_ELEMS + swz(n, k, KT_L)] = (bf16)v;
    }
    // joint fc1 weights: 512 x 1344
    for (int i = g; i < JH_ * 1344; i += stride) {
        int n = i / 1344;
        int k = i - n * 1344;
        wFG[swz(n, k, KT_J)] = (bf16)W_fg[i];
    }
    // output weights padded 29 -> 32 rows with zeros
    for (int i = g; i < 32 * JH_; i += stride) {
        int n = i >> 9;
        int k = i & 511;
        float v = (n < V_) ? W_out[n * JH_ + k] : 0.0f;
        wOUT[swz(n, k, KT_2)] = (bf16)v;
    }
    // embedding table fp32 -> bf16 (plain row-major)
    for (int i = g; i < V_ * H_; i += stride) embb[i] = (bf16)emb[i];
}

// ------------------- LSTM layer: GEMM + fused activation -------------------
// One wave owns jobs (mt, j): output rows mt*16..+16, h-columns j*16..+16 for
// all four gates (i at col, f at +320, g at +640, o at +960).  K-reduction over
// LDS A-tiles; epilogue does sigmoid/tanh in-register per the C/D VGPR layout
// (VGPR r, lanes 0-15 -> M = r, lanes 16-31 -> M = r+8; N = lane%16).
__device__ __forceinline__ void lstm_layer(int l, const bf16* __restrict__ sA,
                                           const bf16* __restrict__ wLl,
                                           const float* __restrict__ b_lstm,
                                           const float* __restrict__ c_ws,
                                           float* __restrict__ hn_ws,
                                           float* __restrict__ cn_ws,
                                           bf16* __restrict__ sA2,
                                           bf16* __restrict__ gAJ,
                                           int wave, int lane)
{
    const int ln15 = lane & 15;
    const int hi   = lane >> 4;
    for (int job = wave; job < 80; job += 32) {
        int mt = job / KT_L;          // 0..3
        int j  = job - mt * KT_L;     // 0..19
        v8f aI = {}, aF = {}, aG = {}, aO = {};
        // per-stream base pointers; K-loop offsets become immediates
        const bf16* aB = sA  + ((mt * KT_L) << 9)                 + (lane << 4);
        const bf16* pI = wLl + (((0 * KT_L + j) * KT_L) << 9)     + (lane << 4);
        const bf16* pF = wLl + (((1 * KT_L + j) * KT_L) << 9)     + (lane << 4);
        const bf16* pG = wLl + (((2 * KT_L + j) * KT_L) << 9)     + (lane << 4);
        const bf16* pO = wLl + (((3 * KT_L + j) * KT_L) << 9)     + (lane << 4);
#pragma unroll
        for (int kt = 0; kt < KT_L; ++kt) {
            v16bf a  = *(const v16bf*)(aB + (kt << 9));
            v16bf bI = *(const v16bf*)(pI + (kt << 9));
            v16bf bF = *(const v16bf*)(pF + (kt << 9));
            v16bf bG = *(const v16bf*)(pG + (kt << 9));
            v16bf bO = *(const v16bf*)(pO + (kt << 9));
            aI = __builtin_amdgcn_wmma_f32_16x16x32_bf16(false, a, false, bI, (short)0, aI, false, false);
            aF = __builtin_amdgcn_wmma_f32_16x16x32_bf16(false, a, false, bF, (short)0, aF, false, false);
            aG = __builtin_amdgcn_wmma_f32_16x16x32_bf16(false, a, false, bG, (short)0, aG, false, false);
            aO = __builtin_amdgcn_wmma_f32_16x16x32_bf16(false, a, false, bO, (short)0, aO, false, false);
        }
        const int hcol = j * 16 + ln15;
        const float bi = b_lstm[l * 1280 + hcol];
        const float bf = b_lstm[l * 1280 + 320 + hcol];
        const float bg = b_lstm[l * 1280 + 640 + hcol];
        const float bo = b_lstm[l * 1280 + 960 + hcol];
#pragma unroll
        for (int r = 0; r < 8; ++r) {
            int m = mt * 16 + r + hi * 8;
            float iv = aI[r] + bi, fv = aF[r] + bf, gv = aG[r] + bg, ov = aO[r] + bo;
            float cp = c_ws[(l * 64 + m) * H_ + hcol];
            float si = 1.0f / (1.0f + __expf(-iv));
            float sf = 1.0f / (1.0f + __expf(-fv));
            float so = 1.0f / (1.0f + __expf(-ov));
            float c2 = sf * cp + si * tanhf(gv);
            float h2 = so * tanhf(c2);
            cn_ws[(l * 64 + m) * H_ + hcol] = c2;
            hn_ws[(l * 64 + m) * H_ + hcol] = h2;
            if (l == 0) sA2[swz(m, hcol, KT_L)] = (bf16)h2;          // layer-2 input
            else        gAJ[swz(m, 1024 + hcol, KT_J)] = (bf16)h2;   // joint input tail
        }
    }
}

// ----------------------------- persistent decoder --------------------------
__launch_bounds__(1024)
__global__ void decode_kernel(const float* __restrict__ x,
                              const int*   __restrict__ x_lens,
                              const float* __restrict__ b_lstm,
                              const float* __restrict__ b1,
                              const float* __restrict__ b_out,
                              const bf16*  __restrict__ wL,
                              const bf16*  __restrict__ wFG,
                              const bf16*  __restrict__ wOUT,
                              const bf16*  __restrict__ embb,
                              bf16*  __restrict__ gAJ,
                              float* __restrict__ h_ws,  float* __restrict__ c_ws,
                              float* __restrict__ hn_ws, float* __restrict__ cn_ws,
                              float* __restrict__ logits_ws,
                              int*   __restrict__ st_ws,
                              int*   __restrict__ out_res,
                              int*   __restrict__ out_len,
                              float* __restrict__ out_h)
{
    extern __shared__ char smem_raw[];
    bf16* sA1  = (bf16*)smem_raw;       // 40960 elems: LSTM-1 A  (64 x 640)
    bf16* sA2  = sA1 + 40960;           // 40960 elems: LSTM-2 A
    bf16* sAF2 = sA1;                   // 32768 elems: fc2 A, aliases sA1

    const int tid  = threadIdx.x;
    const int lane = tid & 31;
    const int wave = tid >> 5;
    const int ln15 = lane & 15;
    const int hi   = lane >> 4;

    int* time_idx = st_ws;
    int* finish   = st_ws + 64;
    int* sym_add  = st_ws + 128;
    int* pred_g   = st_ws + 192;
    int* updg     = st_ws + 256;

    // ---- init state (deterministic every call) ----
    for (int i = tid; i < N_ * MAX_RES_; i += 1024) out_res[i] = -1;
    for (int i = tid; i < 2 * 64 * H_; i += 1024) {
        h_ws[i] = 0.0f; c_ws[i] = 0.0f; hn_ws[i] = 0.0f; cn_ws[i] = 0.0f;
    }
    if (tid < 64) {
        time_idx[tid] = 0;
        finish[tid]   = (x_lens[tid] <= 0) ? 1 : 0;
        sym_add[tid]  = 0;
        pred_g[tid]   = SOS_;
        updg[tid]     = 0;
        out_len[tid]  = 0;
    }
    __syncthreads();

    for (int step = 0; step < MAX_STEPS_; ++step) {
        // ---- P0: pack activation A-tiles ----
        // A1 = [emb[pred] | h_l0]  (64 x 640) -> LDS
        for (int i = tid; i < 64 * 640; i += 1024) {
            int m = i / 640, k = i - m * 640;
            bf16 v = (k < H_) ? embb[pred_g[m] * H_ + k]
                              : (bf16)h_ws[(0 * 64 + m) * H_ + (k - H_)];
            sA1[swz(m, k, KT_L)] = v;
        }
        // A2 upper half = committed h_l1  (k in [320,640))
        for (int i = tid; i < 64 * H_; i += 1024) {
            int m = i / H_, k = i - m * H_;
            sA2[swz(m, H_ + k, KT_L)] = (bf16)h_ws[(1 * 64 + m) * H_ + k];
        }
        // Joint A head = encoder frame f_enc[time_idx[m], m, :]  (k in [0,1024))
        for (int m = 0; m < 64; ++m) {
            int t = time_idx[m];
            gAJ[swz(m, tid, KT_J)] = (bf16)x[((size_t)t * 64 + m) * C_ + tid];
        }
        __syncthreads();

        // ---- P1: LSTM layer 0 (GEMM + fused gate activation) ----
        lstm_layer(0, sA1, wL,            b_lstm, c_ws, hn_ws, cn_ws, sA2, gAJ, wave, lane);
        __syncthreads();

        // ---- P2: LSTM layer 1 ----
        lstm_layer(1, sA2, wL + WL_ELEMS, b_lstm, c_ws, hn_ws, cn_ws, sA2, gAJ, wave, lane);
        __syncthreads();

        // ---- P3: joint fc1 (64x512, K=1344) + ReLU, pack fc2 A into LDS ----
        for (int job = wave; job < 128; job += 32) {
            int mt = job >> 5, nt = job & 31;
            v8f acc = {};
            const bf16* aB = gAJ + ((mt * KT_J) << 9) + (lane << 4);
            const bf16* bB = wFG + ((nt * KT_J) << 9) + (lane << 4);
#pragma unroll 7
            for (int kt = 0; kt < KT_J; ++kt) {
                v16bf a = *(const v16bf*)(aB + (kt << 9));
                v16bf b = *(const v16bf*)(bB + (kt << 9));
                acc = __builtin_amdgcn_wmma_f32_16x16x32_bf16(false, a, false, b, (short)0, acc, false, false);
            }
            int col = nt * 16 + ln15;
            float bias = b1[col];
#pragma unroll
            for (int r = 0; r < 8; ++r) {
                int m = mt * 16 + r + hi * 8;
                float v = acc[r] + bias;
                v = v > 0.0f ? v : 0.0f;
                sAF2[swz(m, col, KT_2)] = (bf16)v;
            }
        }
        __syncthreads();

        // ---- P4: fc2 (64x32, K=512); cols >=29 biased to -1e30 ----
        if (wave < 8) {
            int mt = wave >> 1, nt = wave & 1;
            v8f acc = {};
            const bf16* aB = sAF2 + ((mt * KT_2) << 9) + (lane << 4);
            const bf16* bB = wOUT + ((nt * KT_2) << 9) + (lane << 4);
#pragma unroll
            for (int kt = 0; kt < KT_2; ++kt) {
                v16bf a = *(const v16bf*)(aB + (kt << 9));
                v16bf b = *(const v16bf*)(bB + (kt << 9));
                acc = __builtin_amdgcn_wmma_f32_16x16x32_bf16(false, a, false, b, (short)0, acc, false, false);
            }
            int col = nt * 16 + ln15;
            float bias = (col < V_) ? b_out[col] : -1e30f;
#pragma unroll
            for (int r = 0; r < 8; ++r) {
                int m = mt * 16 + r + hi * 8;
                logits_ws[m * 32 + col] = acc[r] + bias;
            }
        }
        __syncthreads();

        // ---- P5a: per-row argmax + state machine (reference semantics) ----
        if (tid < 64) {
            int m = tid;
            float best = -3.4e38f;
            int sym = 0;
            for (int c = 0; c < V_; ++c) {
                float v = logits_ws[m * 32 + c];
                if (v > best) { best = v; sym = c; }
            }
            int fin = finish[m], sa = sym_add[m];
            int upd = (sym != BLANK_) && (sa < MAX_SYM_) && !fin;
            if (upd) {
                int rl = out_len[m];
                out_res[m * MAX_RES_ + rl] = sym;
                out_len[m] = rl + 1;
                sa += 1;
                pred_g[m] = sym;
            }
            int updf = (!upd) && (!fin);
            int ti = time_idx[m] + (updf ? 1 : 0);
            int lenm = x_lens[m];
            int fin2 = fin || (ti >= lenm);
            ti = ti < (lenm - 1) ? ti : (lenm - 1);
            if (ti < 0) ti = 0;
            if (updf) sa = 0;
            time_idx[m] = ti;
            finish[m]   = fin2;
            sym_add[m]  = sa;
            updg[m]     = upd;
        }
        __syncthreads();

        // ---- P5b: conditional commit of h/c ----
        for (int i = tid; i < 2 * 64 * H_; i += 1024) {
            int m = (i / H_) & 63;
            if (updg[m]) { h_ws[i] = hn_ws[i]; c_ws[i] = cn_ws[i]; }
        }
        __syncthreads();
    }

    // ---- emit committed hidden state ----
    for (int i = tid; i < 2 * 64 * H_; i += 1024) out_h[i] = h_ws[i];
}

// ----------------------------------- launch --------------------------------
extern "C" void kernel_launch(void* const* d_in, const int* in_sizes, int n_in,
                              void* d_out, int out_size, void* d_ws, size_t ws_size,
                              hipStream_t stream) {
    const float* x      = (const float*)d_in[0];
    const int*   x_lens = (const int*)  d_in[1];
    const float* emb    = (const float*)d_in[2];
    const float* W_ih   = (const float*)d_in[3];
    const float* W_hh   = (const float*)d_in[4];
    const float* b_lstm = (const float*)d_in[5];
    const float* W_fg   = (const float*)d_in[6];
    const float* b1     = (const float*)d_in[7];
    const float* W_out  = (const float*)d_in[8];
    const float* b_out  = (const float*)d_in[9];

    // workspace layout (256B aligned)
    char* ws = (char*)d_ws;
    size_t off = 0;
    auto take = [&](size_t bytes) { char* p = ws + off; off = (off + bytes + 255) & ~(size_t)255; return p; };
    bf16*  wL     = (bf16*) take(2 * WL_ELEMS * sizeof(bf16));       // 3.28 MB
    bf16*  wFG    = (bf16*) take(32 * KT_J * 512 * sizeof(bf16));    // 1.38 MB
    bf16*  wOUT   = (bf16*) take(2 * KT_2 * 512 * sizeof(bf16));     // 32 KB
    bf16*  embb   = (bf16*) take(V_ * H_ * sizeof(bf16));
    bf16*  gAJ    = (bf16*) take(4 * KT_J * 512 * sizeof(bf16));     // 172 KB
    float* h_ws   = (float*)take(2 * 64 * H_ * sizeof(float));
    float* c_ws   = (float*)take(2 * 64 * H_ * sizeof(float));
    float* hn_ws  = (float*)take(2 * 64 * H_ * sizeof(float));
    float* cn_ws  = (float*)take(2 * 64 * H_ * sizeof(float));
    float* logits = (float*)take(64 * 32 * sizeof(float));
    int*   st_ws  = (int*)  take(5 * 64 * sizeof(int));
    (void)ws_size; (void)in_sizes; (void)n_in; (void)out_size;

    // outputs: res_buf (int 64*601) | res_len (int 64) | h (f32 2*64*320)
    int*   out_res = (int*)d_out;
    int*   out_len = out_res + N_ * MAX_RES_;
    float* out_h   = (float*)d_out + (N_ * MAX_RES_ + N_);

    prep_kernel<<<1024, 256, 0, stream>>>(W_ih, W_hh, W_fg, W_out, emb,
                                          wL, wFG, wOUT, embb);

    const size_t smem = (40960 + 40960) * sizeof(bf16);  // 160 KB of WGP LDS
    decode_kernel<<<1, 1024, smem, stream>>>(x, x_lens, b_lstm, b1, b_out,
                                             wL, wFG, wOUT, embb, gAJ,
                                             h_ws, c_ws, hn_ws, cn_ws, logits,
                                             st_ws, out_res, out_len, out_h);
}
